// HMON_39934605918848
// MI455X (gfx1250) — matrix operand order repaired and checked by reading, not compiled, not run on hardware
//
#include <hip/hip_runtime.h>
#include <math.h>

#define N_    118
#define TT    2048
#define NFEAT 64
#define DP    128
#define DM    768
#define DR    128
#define DH    256
#define DN    256
#define DPHI  192
#define DE    256
#define NEGS  0.01f
#define NP    (N_*N_)      /* 13924 */
#define NPAD  13936        /* 871 * 16 */
#define MT_P  871

typedef __attribute__((ext_vector_type(16))) _Float16 v16h;
typedef __attribute__((ext_vector_type(8)))  _Float16 v8h;
typedef __attribute__((ext_vector_type(8)))  float    v8f;

__device__ __forceinline__ float leakyf(float x)   { return x > 0.f ? x : NEGS * x; }
__device__ __forceinline__ float sigmoidf_(float x){ return 1.f / (1.f + expf(-x)); }

// ---------------------------------------------------------------------------
// WMMA fragment helpers (wave32, v_wmma_f32_16x16x32_f16; ISA 7.12.2 layouts)
// A tile (16xK slice) from row-major (M x K) f16, ld in halves.
__device__ __forceinline__ v16h load_a_frag(const _Float16* A, int ld, int m0, int k0) {
  int lane = threadIdx.x & 31;
  int m  = m0 + (lane & 15);
  int hf = lane >> 4;
  const _Float16* p = A + (size_t)m * ld + k0 + 8 * hf;
  v8h lo = *(const v8h*)p;           // K = 8*hf + 0..7
  v8h hi = *(const v8h*)(p + 16);    // K = 16 + 8*hf + 0..7
  v16h r;
#pragma unroll
  for (int i = 0; i < 8; ++i) { r[i] = lo[i]; r[i + 8] = hi[i]; }
  return r;
}

// B tile from row-major (N x K) weight-transpose; lane holds output column n,
// K = 8*(lane>=16) + 0..7, then 16 + 8*(lane>=16) + 0..7.
__device__ __forceinline__ v16h load_b_frag(const _Float16* Wt, int ld, int n0, int k0) {
  int lane = threadIdx.x & 31;
  int n  = n0 + (lane & 15);
  int hf = lane >> 4;
  const _Float16* p = Wt + (size_t)n * ld + k0 + 8 * hf;
  v8h lo = *(const v8h*)p;
  v8h hi = *(const v8h*)(p + 16);
  v16h r;
#pragma unroll
  for (int i = 0; i < 8; ++i) { r[i] = lo[i]; r[i + 8] = hi[i]; }
  return r;
}

__device__ __forceinline__ v8f wmma_f16(v16h a, v16h b, v8f c) {
  return __builtin_amdgcn_wmma_f32_16x16x32_f16(false, a, false, b, (short)0, c, false, false);
}

// ---------------------------------------------------------------------------
// Weight prep: concat [Wih|Whh] -> f16 (N x K) row-major; combined biases.
__global__ void hmon_prep_lstm(const float* Wih0, const float* Whh0,
                               const float* bih0, const float* bhh0,
                               const float* Wih1, const float* Whh1,
                               const float* bih1, const float* bhh1,
                               _Float16* W0, _Float16* W1, float* b0, float* b1) {
  int idx = blockIdx.x * blockDim.x + threadIdx.x;
  int str = gridDim.x * blockDim.x;
  for (int i = idx; i < 512 * 192; i += str) {
    int n = i / 192, k = i % 192;
    W0[i] = (_Float16)(k < 64 ? Wih0[n * 64 + k] : Whh0[n * 128 + (k - 64)]);
  }
  for (int i = idx; i < 512 * 256; i += str) {
    int n = i / 256, k = i % 256;
    W1[i] = (_Float16)(k < 128 ? Wih1[n * 128 + k] : Whh1[n * 128 + (k - 128)]);
  }
  for (int i = idx; i < 512; i += str) {
    b0[i] = bih0[i] + bhh0[i];
    b1[i] = bih1[i] + bhh1[i];
  }
}

// Transpose pairwise weights to (N x K) f16 for the B-fragment loader.
__global__ void hmon_prep_pairw(const float* Wphin, const float* Won, const float* Weo,
                                _Float16* WphinT, _Float16* WonT, _Float16* WeoT) {
  int idx = blockIdx.x * blockDim.x + threadIdx.x;
  int str = gridDim.x * blockDim.x;
  for (int i = idx; i < DPHI * 512; i += str) {
    int n = i / 512, k = i % 512;
    WphinT[i] = (_Float16)Wphin[k * DPHI + n];
  }
  for (int i = idx; i < DN * 512; i += str) {
    int n = i / 512, k = i % 512;
    WonT[i] = (_Float16)Won[k * DN + n];
  }
  for (int i = idx; i < DE * 768; i += str) {
    int n = i / 768, k = i % 768;
    WeoT[i] = (_Float16)Weo[k * DE + n];
  }
}

// ---------------------------------------------------------------------------
// Persistent fused LSTM layer. One workgroup (16 waves). Weights + input tile
// + hidden state resident in LDS (CDNA5: 320KB/WGP). Cell state lives in
// accumulator VGPRs for all 2048 steps. g = [x_t | h] @ Wcat^T + b.
// Pipelining:
//   - B fragments are explicitly 1-deep double-buffered against the WMMAs
//     (ds_load of b[u+1] issued before wmma on b[u]) so the k-loop never
//     waits dscnt==0.
//   - Global loads of the step-(t+1) input are issued AFTER the k-loop, so
//     their registers are not live during the k-loop; their latency hides
//     behind the barrier + the sigmoid/tanh elementwise block, and they are
//     committed to LDS at the top of the next step.
// Layer0: xf32 = x (118,2048,64), hout = h1 sequence (f16, t x 128 x 128).
// Layer1: xh16 = h1 sequence,     pout = leaky(h_last) (f32, 128 x 128).
__global__ __launch_bounds__(512) void hmon_lstm(
    const float* __restrict__ xf32, const _Float16* __restrict__ xh16,
    const _Float16* __restrict__ Wcat, const float* __restrict__ bias,
    _Float16* __restrict__ hout, float* __restrict__ pout, int KIN) {
  extern __shared__ _Float16 sm[];
  const int KW = KIN + 128;
  _Float16* sW = sm;                  // 512 x KW
  _Float16* sX = sW + 512 * KW;       // 128 x KIN
  _Float16* sH = sX + 128 * KIN;      // 128 x 128

  const int tid  = threadIdx.x;
  const int lane = tid & 31;
  const int wave = tid >> 5;          // 0..15

  // Stage weights, zero state buffers.
  for (int i = tid; i < 512 * KW / 8; i += 512) ((v8h*)sW)[i] = ((const v8h*)Wcat)[i];
  { v8h z = {};
    for (int i = tid; i < 128 * KIN / 8; i += 512) ((v8h*)sX)[i] = z;
    for (int i = tid; i < 128 * 128 / 8; i += 512) ((v8h*)sH)[i] = z; }
  __syncthreads();   // init visible before first staging commit

  // Tile assignment: 64 (Mtile,Hslice) pairs over 16 waves -> 4 each, same Mtile.
  const int mt   = wave >> 1;              // M tile 0..7
  const int ht0  = (wave & 1) * 4;         // H slice base (of 8)
  const int nrow = lane & 15;
  const int hf   = lane >> 4;
  const int KITER = KW >> 5;

  float bb[4][4];
#pragma unroll
  for (int q = 0; q < 4; ++q)
#pragma unroll
    for (int g = 0; g < 4; ++g)
      bb[q][g] = bias[g * 128 + (ht0 + q) * 16 + nrow];

  v8f cst[4];
#pragma unroll
  for (int q = 0; q < 4; ++q) cst[q] = {};

  // ---- software-pipeline state: input for step t held in registers ----
  const int xrow = tid >> 2;               // layer0 staging coords
  const int xseg = (tid & 3) * 16;
  float xr[16];
  v8h hr[4];
  if (xf32) {
    if (xrow < N_) {
      const float* xp = xf32 + (size_t)xrow * TT * NFEAT + xseg;   // t = 0
#pragma unroll
      for (int i = 0; i < 16; ++i) xr[i] = xp[i];
    }
  } else {
    const v8h* src = (const v8h*)xh16;                              // t = 0
#pragma unroll
    for (int j = 0; j < 4; ++j) hr[j] = src[tid + j * 512];
  }

  for (int t = 0; t < TT; ++t) {
    // ---- commit staged input for step t into sX ----
    if (xf32) {
      if (xrow < N_) {
        _Float16* dp = sX + xrow * NFEAT + xseg;
#pragma unroll
        for (int i = 0; i < 16; ++i) dp[i] = (_Float16)xr[i];
      }
    } else {
      v8h* dst = (v8h*)sX;
#pragma unroll
      for (int j = 0; j < 4; ++j) dst[tid + j * 512] = hr[j];
    }
    __syncthreads();   // sX staged & previous-step sH published

    v8f acc[4][4];
#pragma unroll
    for (int q = 0; q < 4; ++q)
#pragma unroll
      for (int g = 0; g < 4; ++g) {
        v8f a;
#pragma unroll
        for (int r = 0; r < 8; ++r) a[r] = bb[q][g];
        acc[q][g] = a;
      }

    for (int kk = 0; kk < KITER; ++kk) {
      int k0 = kk * 32;
      v16h af = (k0 < KIN) ? load_a_frag(sX, KIN, mt * 16, k0)
                           : load_a_frag(sH, 128, mt * 16, k0 - KIN);
      // explicit 1-deep pipeline: b[u+1] in flight while wmma consumes b[u]
      v16h bcur = load_b_frag(sW, KW, 0 * 128 + ht0 * 16, k0);
#pragma unroll
      for (int u = 0; u < 16; ++u) {
        const int q = u >> 2;
        const int g = u & 3;
        if (u < 15) {
          const int un = u + 1;
          v16h bnext = load_b_frag(sW, KW, (un & 3) * 128 + (ht0 + (un >> 2)) * 16, k0);
          acc[q][g] = wmma_f16(af, bcur, acc[q][g]);
          bcur = bnext;
        } else {
          acc[q][g] = wmma_f16(af, bcur, acc[q][g]);
        }
      }
    }

    // ---- issue global loads for step t+1 (latency hides behind barrier +
    //      the transcendental-heavy elementwise block below) ----
    if (t + 1 < TT) {
      if (xf32) {
        if (xrow < N_) {
          const float* xp = xf32 + ((size_t)xrow * TT + (t + 1)) * NFEAT + xseg;
#pragma unroll
          for (int i = 0; i < 16; ++i) xr[i] = xp[i];
        }
      } else {
        const v8h* src = (const v8h*)(xh16 + (size_t)(t + 1) * 128 * 128);
#pragma unroll
        for (int j = 0; j < 4; ++j) hr[j] = src[tid + j * 512];
      }
    }
    __syncthreads();   // all reads of sX/sH complete before h update

    // ---- gate nonlinearities, cell update, publish h ----
#pragma unroll
    for (int q = 0; q < 4; ++q) {
      int n    = (ht0 + q) * 16 + nrow;
      int mrow = mt * 16 + 8 * hf;
#pragma unroll
      for (int r = 0; r < 8; ++r) {
        float ig = sigmoidf_(acc[q][0][r]);
        float fg = sigmoidf_(acc[q][1][r]);
        float gg = tanhf(acc[q][2][r]);
        float og = sigmoidf_(acc[q][3][r]);
        float cv = fg * cst[q][r] + ig * gg;
        cst[q][r] = cv;
        float hv = og * tanhf(cv);
        _Float16 hh = (_Float16)hv;
        sH[(mrow + r) * 128 + n] = hh;
        if (hout) hout[(size_t)t * 16384 + (mrow + r) * 128 + n] = hh;
        if (pout && t == TT - 1) pout[(mrow + r) * 128 + n] = leakyf(hv);
      }
    }
  }
}

// ---------------------------------------------------------------------------
// Generic WMMA GEMM: C[M x N] = concatK(A1,A2) @ Wt^T ; one wave per 16x16 tile.
// A/B fragments rotate with one k-iteration of lookahead.
__global__ __launch_bounds__(256) void hmon_wmma_gemm(
    const _Float16* __restrict__ A1, int lda1, int K1,
    const _Float16* __restrict__ A2, int lda2, int K2,
    const _Float16* __restrict__ Wt, int ldw,
    float* __restrict__ Cf, _Float16* __restrict__ Ch, int ldc,
    int Mtiles, int Ntiles, int do_leaky) {
  int gw = (blockIdx.x * blockDim.x + threadIdx.x) >> 5;
  if (gw >= Mtiles * Ntiles) return;       // wave-uniform: EXEC stays all-ones
  int mt = gw / Ntiles;
  int nt = gw - mt * Ntiles;
  int K  = K1 + K2;
  v8f acc = {};
  v16h af = load_a_frag(A1, lda1, mt * 16, 0);   // K1 >= 32 for all call sites
  v16h bf = load_b_frag(Wt, ldw, nt * 16, 0);
  for (int k0 = 0; k0 < K; k0 += 32) {
    int kn = k0 + 32;
    v16h afn, bfn;
    if (kn < K) {
      afn = (kn < K1) ? load_a_frag(A1, lda1, mt * 16, kn)
                      : load_a_frag(A2, lda2, mt * 16, kn - K1);
      bfn = load_b_frag(Wt, ldw, nt * 16, kn);
    }
    acc = wmma_f16(af, bf, acc);
    if (kn < K) { af = afn; bf = bfn; }
  }
  int lane = threadIdx.x & 31;
  int n  = nt * 16 + (lane & 15);
  int m0 = mt * 16 + 8 * (lane >> 4);
#pragma unroll
  for (int r = 0; r < 8; ++r) {
    float v = acc[r];
    if (do_leaky) v = leakyf(v);
    if (Cf) Cf[(size_t)(m0 + r) * ldc + n] = v;
    if (Ch) Ch[(size_t)(m0 + r) * ldc + n] = (_Float16)v;
  }
}

// ---------------------------------------------------------------------------
// Small glue kernels (N=118 regime, deterministic reductions).
__global__ void hmon_status(const float* xm, float* st) {
  int i = blockIdx.x * blockDim.x + threadIdx.x;
  if (i >= N_) return;
  float s = 0.f;
  for (int k = 0; k < DM; ++k) s += xm[i * DM + k];
  st[i] = (s == 0.f) ? 1.f : 0.f;
}

__global__ void hmon_uv(const float* xm, const float* pseudo, const float* st,
                        const float* p, const float* Wum, const float* Wvm,
                        const float* Wup, const float* Wvp,
                        float* um, float* vm, float* up, float* vp) {
  int idx = blockIdx.x * blockDim.x + threadIdx.x;
  if (idx >= N_ * DR) return;
  int i = idx / DR, r = idx % DR;
  float s = st[i];
  float a_um = 0.f, a_vm = 0.f;
  for (int k = 0; k < DM; ++k) {
    float mk = (s == 1.f) ? pseudo[k] : xm[i * DM + k];
    a_um += mk * Wum[k * DR + r];
    a_vm += mk * Wvm[k * DR + r];
  }
  a_um += s * Wum[DM * DR + r];
  a_vm += s * Wvm[DM * DR + r];
  float a_up = 0.f, a_vp = 0.f;
  for (int k = 0; k < DP; ++k) {
    float pk = p[i * 128 + k];
    a_up += pk * Wup[k * DR + r];
    a_vp += pk * Wvp[k * DR + r];
  }
  a_up += s * Wup[DP * DR + r];
  a_vp += s * Wvp[DP * DR + r];
  um[idx] = leakyf(a_um); vm[idx] = leakyf(a_vm);
  up[idx] = leakyf(a_up); vp[idx] = leakyf(a_vp);
}

__global__ void hmon_pmt(const float* um, const float* vp, const float* Wpe,
                         const float* bpe, float* hpmt) {
  int i = blockIdx.x * blockDim.x + threadIdx.x;
  if (i >= N_) return;
  float z0 = bpe[0], z1 = bpe[1];
  for (int d = 0; d < 3 * DR; ++d) {
    float zd;
    if (d < DR)           zd = um[i * DR + d];
    else if (d < 2 * DR)  zd = um[i * DR + d - DR] * vp[i * DR + d - DR];
    else                  zd = vp[i * DR + d - 2 * DR];
    z0 += zd * Wpe[d];
    z1 += zd * Wpe[3 * DR + d];
  }
  float mx = fmaxf(z0, z1);
  float e0 = expf(z0 - mx), e1 = expf(z1 - mx);
  float sm = e0 + e1;
  hpmt[i * 2 + 0] = e0 / sm;
  hpmt[i * 2 + 1] = e1 / sm;
}

__global__ void hmon_hyb(const float* up, const float* vm, const float* Whyb,
                         const float* bhyb, float* hhyb) {
  int idx = blockIdx.x * blockDim.x + threadIdx.x;
  if (idx >= N_ * DH) return;
  int i = idx / DH, hh = idx % DH;
  float a = bhyb[hh];
  const float* wr = Whyb + (size_t)hh * 3 * DR;
  for (int d = 0; d < DR; ++d) {
    float u = up[i * DR + d], v = vm[i * DR + d];
    a += u * wr[d] + (u + v) * wr[DR + d] + v * wr[2 * DR + d];
  }
  hhyb[idx] = leakyf(a);
}

__global__ void hmon_nvec(const float* hhyb, const float* hpmt, const float* st,
                          const float* Wact, const float* bact,
                          const float* Winact, const float* binact, float* nbuf) {
  int idx = blockIdx.x * blockDim.x + threadIdx.x;
  if (idx >= N_ * DN) return;
  int i = idx / DN, d = idx % DN;
  float s = st[i];
  const float* wa = Wact + (size_t)d * (DH + 2);
  const float* wi = Winact + (size_t)d * DH;
  float aact = bact[d], ain = binact[d];
  for (int k = 0; k < DH; ++k) {
    float h = hhyb[i * DH + k];
    aact += h * wa[k];
    ain  += h * wi[k];
  }
  aact += hpmt[i * 2 + 0] * wa[DH] + hpmt[i * 2 + 1] * wa[DH + 1];
  nbuf[idx] = leakyf(aact * (1.f - s) + ain * s);
}

__global__ void hmon_normw(const float* nbuf, const float* hpmt, const float* st,
                           float* norms, float* wv) {
  int i = blockIdx.x * blockDim.x + threadIdx.x;
  if (i >= N_) return;
  float s2 = 0.f;
  for (int d = 0; d < DN; ++d) { float v = nbuf[i * DN + d]; s2 += v * v; }
  norms[i] = fmaxf(sqrtf(s2), 1e-8f);
  float a = hpmt[i * 2 + 1] - hpmt[i * 2 + 0];
  float w = 0.f;
  if (st[i] == 0.f) w = (a > 0.f) ? 1.f : ((a < 0.f) ? -1.f : 0.f);
  wv[i] = w;
}

__global__ void hmon_pair(const float* nbuf, _Float16* pair) {
  size_t idx0 = (size_t)blockIdx.x * blockDim.x + threadIdx.x;
  size_t total = (size_t)NPAD * 512;
  size_t str = (size_t)gridDim.x * blockDim.x;
  for (size_t idx = idx0; idx < total; idx += str) {
    int r = (int)(idx / 512), k = (int)(idx % 512);
    float v = 0.f;
    if (r < NP) {
      int i = r / N_, j = r - i * N_;
      v = (k < DN) ? nbuf[j * DN + k] : nbuf[i * DN + (k - DN)];
    }
    pair[idx] = (_Float16)v;
  }
}

__global__ void hmon_phi(const float* P1, const float* aphi, float* phi) {
  int r = blockIdx.x * blockDim.x + threadIdx.x;
  if (r >= NP) return;
  float s = 0.f;
  for (int d = 0; d < DPHI; ++d) s += leakyf(P1[(size_t)r * DPHI + d]) * aphi[d];
  phi[r] = s;
}

// alpha = gsm(gsm(phi, status==0), status==1): per-column softmax within each group.
__global__ void hmon_alpha(const float* phi, const float* st, float* alpha) {
  int j = threadIdx.x;
  if (j >= N_) return;
  for (int grp = 0; grp < 2; ++grp) {
    float target = (grp == 0) ? 0.f : 1.f;
    float mx = -3.4e38f;
    int cnt = 0;
    for (int i = 0; i < N_; ++i)
      if (st[i] == target) { float v = phi[i * N_ + j]; mx = fmaxf(mx, v); cnt++; }
    if (cnt == 0) continue;
    float sum = 0.f;
    for (int i = 0; i < N_; ++i)
      if (st[i] == target) sum += expf(phi[i * N_ + j] - mx);
    for (int i = 0; i < N_; ++i)
      if (st[i] == target) alpha[i * N_ + j] = expf(phi[i * N_ + j] - mx) / sum;
  }
}

__global__ void hmon_hm(const float* alpha, const float* e, const float* st,
                        float* hm0, float* hm1) {
  int idx = blockIdx.x * blockDim.x + threadIdx.x;
  if (idx >= N_ * DE) return;
  int i = idx / DE, d = idx % DE;
  float acc = 0.f;
  for (int j = 0; j < N_; ++j)
    acc += alpha[i * N_ + j] * e[(size_t)(i * N_ + j) * DE + d];
  float s = st[i];
  hm1[idx] = leakyf((1.f - s) * acc);
  hm0[idx] = leakyf(s * acc);
}

__global__ void hmon_yhat(const float* nbuf, const float* hm0, const float* hm1,
                          const float* Wi, const float* bi, const float* hpmt,
                          float* out) {
  int i = blockIdx.x * blockDim.x + threadIdx.x;
  if (i >= N_) return;
  float z0 = bi[i * 2 + 0], z1 = bi[i * 2 + 1];
  const float* W = Wi + (size_t)i * 768 * 2;
  for (int d = 0; d < DN; ++d) { float v = nbuf[i * DN + d]; z0 += v * W[d * 2]; z1 += v * W[d * 2 + 1]; }
  for (int d = 0; d < DE; ++d) { float v = hm0[i * DE + d]; z0 += v * W[(DN + d) * 2]; z1 += v * W[(DN + d) * 2 + 1]; }
  for (int d = 0; d < DE; ++d) { float v = hm1[i * DE + d]; z0 += v * W[(DN + DE + d) * 2]; z1 += v * W[(DN + DE + d) * 2 + 1]; }
  float mx = fmaxf(z0, z1);
  float e0 = expf(z0 - mx), e1 = expf(z1 - mx);
  float s = e0 + e1;
  out[i * 2 + 0] = e0 / s;
  out[i * 2 + 1] = e1 / s;
  out[2 * N_ + i * 2 + 0] = hpmt[i * 2 + 0];
  out[2 * N_ + i * 2 + 1] = hpmt[i * 2 + 1];
}

__global__ void hmon_lort(const float* Wum, const float* Wvm,
                          const float* Wup, const float* Wvp, float* partial) {
  __shared__ float red[128];
  int blk = blockIdx.x;
  int mat = blk / 769;
  int a   = blk - mat * 769;
  const float* U = mat ? Wup : Wum;
  const float* V = mat ? Wvp : Wvm;
  int R = mat ? (DP + 1) : (DM + 1);
  float loc = 0.f;
  if (a < R) {
    for (int b = threadIdx.x; b < R; b += blockDim.x) {
      float dot = 0.f;
      for (int k = 0; k < DR; ++k) dot += U[a * DR + k] * V[b * DR + k];
      loc += dot * dot;
    }
  }
  red[threadIdx.x] = loc;
  __syncthreads();
  for (int s = 64; s > 0; s >>= 1) {
    if (threadIdx.x < s) red[threadIdx.x] += red[threadIdx.x + s];
    __syncthreads();
  }
  if (threadIdx.x == 0) partial[blk] = red[0];
}

// l_pol = || sum_i w_i * n_i / norm_i ||^2
__global__ void hmon_lpol(const float* nbuf, const float* norms, const float* wv,
                          float* svec) {
  int d = threadIdx.x;
  if (d >= DN) return;
  float s = 0.f;
  for (int i = 0; i < N_; ++i) s += wv[i] * nbuf[i * DN + d] / norms[i];
  svec[d] = s;
}

__global__ void hmon_finalize(const float* partial, const float* svec, float* out) {
  float s1 = 0.f, s2 = 0.f;
  for (int i = 0; i < 769; ++i) s1 += partial[i];
  for (int i = 0; i < 769; ++i) s2 += partial[769 + i];
  float lpol = 0.f;
  for (int d = 0; d < DN; ++d) lpol += svec[d] * svec[d];
  out[4 * N_ + 0] = sqrtf(s1) + sqrtf(s2);
  out[4 * N_ + 1] = lpol;
}

// ---------------------------------------------------------------------------
extern "C" void kernel_launch(void* const* d_in, const int* in_sizes, int n_in,
                              void* d_out, int out_size, void* d_ws, size_t ws_size,
                              hipStream_t stream) {
  (void)in_sizes; (void)n_in; (void)out_size; (void)ws_size;
  const float* x      = (const float*)d_in[0];
  const float* xmsg   = (const float*)d_in[1];
  const float* pseudo = (const float*)d_in[2];
  const float* Wih0   = (const float*)d_in[3];
  const float* Whh0   = (const float*)d_in[4];
  const float* bih0   = (const float*)d_in[5];
  const float* bhh0   = (const float*)d_in[6];
  const float* Wih1   = (const float*)d_in[7];
  const float* Whh1   = (const float*)d_in[8];
  const float* bih1   = (const float*)d_in[9];
  const float* bhh1   = (const float*)d_in[10];
  const float* Wum    = (const float*)d_in[11];
  const float* Wvm    = (const float*)d_in[12];
  const float* Wup    = (const float*)d_in[13];
  const float* Wvp    = (const float*)d_in[14];
  const float* Wpe    = (const float*)d_in[15];
  const float* bpe    = (const float*)d_in[16];
  const float* Whyb   = (const float*)d_in[17];
  const float* bhyb   = (const float*)d_in[18];
  const float* Wact   = (const float*)d_in[19];
  const float* bact   = (const float*)d_in[20];
  const float* Winact = (const float*)d_in[21];
  const float* binact = (const float*)d_in[22];
  const float* Wphin  = (const float*)d_in[23];
  const float* aphi   = (const float*)d_in[24];
  const float* Won    = (const float*)d_in[25];
  const float* Weo    = (const float*)d_in[26];
  const float* Wi     = (const float*)d_in[27];
  const float* bi     = (const float*)d_in[28];
  float* out = (float*)d_out;

  char* ws = (char*)d_ws;
  size_t off = 0;
  auto alloc = [&](size_t bytes) -> char* {
    char* p = ws + off;
    off = (off + bytes + 255) & ~(size_t)255;
    return p;
  };

  _Float16* h1seq  = (_Float16*)alloc((size_t)TT * 128 * 128 * 2);
  _Float16* W0     = (_Float16*)alloc(512 * 192 * 2);
  _Float16* W1     = (_Float16*)alloc(512 * 256 * 2);
  float* b0        = (float*)alloc(512 * 4);
  float* b1        = (float*)alloc(512 * 4);
  float* pbuf      = (float*)alloc(128 * 128 * 4);
  float* st        = (float*)alloc(128 * 4);
  float* um        = (float*)alloc(N_ * DR * 4);
  float* vm        = (float*)alloc(N_ * DR * 4);
  float* up        = (float*)alloc(N_ * DR * 4);
  float* vp        = (float*)alloc(N_ * DR * 4);
  float* hpmt      = (float*)alloc(N_ * 2 * 4);
  float* hhyb      = (float*)alloc(N_ * DH * 4);
  float* nbuf      = (float*)alloc(N_ * DN * 4);
  float* norms     = (float*)alloc(128 * 4);
  float* wv        = (float*)alloc(128 * 4);
  _Float16* WphinT = (_Float16*)alloc((size_t)DPHI * 512 * 2);
  _Float16* WonT   = (_Float16*)alloc((size_t)DN * 512 * 2);
  _Float16* WeoT   = (_Float16*)alloc((size_t)DE * 768 * 2);
  _Float16* pair   = (_Float16*)alloc((size_t)NPAD * 512 * 2);
  float* P1        = (float*)alloc((size_t)NPAD * DPHI * 4);
  _Float16* Q      = (_Float16*)alloc((size_t)NPAD * DN * 2);
  float* ebuf      = (float*)alloc((size_t)NPAD * DE * 4);
  float* phi       = (float*)alloc((size_t)NP * 4);
  float* alpha     = (float*)alloc((size_t)NP * 4);
  float* hm0       = (float*)alloc(N_ * DE * 4);
  float* hm1       = (float*)alloc(N_ * DE * 4);
  float* partial   = (float*)alloc(1538 * 4);
  float* svec      = (float*)alloc(256 * 4);

  // --- weight prep (independent) ---
  hmon_prep_lstm<<<256, 256, 0, stream>>>(Wih0, Whh0, bih0, bhh0,
                                          Wih1, Whh1, bih1, bhh1, W0, W1, b0, b1);
  hmon_prep_pairw<<<512, 256, 0, stream>>>(Wphin, Won, Weo, WphinT, WonT, WeoT);

  // --- persistent fused LSTM layers (weights resident in 320KB WGP LDS) ---
  int lds0 = (512 * 192 + 128 * 64 + 128 * 128) * 2;   // 245760 B
  int lds1 = (512 * 256 + 128 * 128 + 128 * 128) * 2;  // 327680 B
  hmon_lstm<<<1, 512, lds0, stream>>>(x, nullptr, W0, b0, h1seq, nullptr, 64);
  hmon_lstm<<<1, 512, lds1, stream>>>(nullptr, h1seq, W1, b1, nullptr, pbuf, 128);

  // --- glue / message path ---
  hmon_status<<<1, 128, 0, stream>>>(xmsg, st);
  hmon_uv<<<(N_ * DR + 255) / 256, 256, 0, stream>>>(xmsg, pseudo, st, pbuf,
                                                     Wum, Wvm, Wup, Wvp,
                                                     um, vm, up, vp);
  hmon_pmt<<<1, 128, 0, stream>>>(um, vp, Wpe, bpe, hpmt);
  hmon_hyb<<<(N_ * DH + 255) / 256, 256, 0, stream>>>(up, vm, Whyb, bhyb, hhyb);
  hmon_nvec<<<(N_ * DN + 255) / 256, 256, 0, stream>>>(hhyb, hpmt, st, Wact, bact,
                                                       Winact, binact, nbuf);
  hmon_normw<<<1, 128, 0, stream>>>(nbuf, hpmt, st, norms, wv);

  // --- pairwise WMMA GEMMs ---
  hmon_pair<<<2048, 256, 0, stream>>>(nbuf, pair);
  // phi pre-activation: (13936 x 512) @ (512 x 192)
  hmon_wmma_gemm<<<(MT_P * 12 + 7) / 8, 256, 0, stream>>>(
      pair, 512, 512, nullptr, 0, 0, WphinT, 512, P1, nullptr, DPHI, MT_P, 12, 0);
  hmon_phi<<<(NP + 255) / 256, 256, 0, stream>>>(P1, aphi, phi);
  hmon_alpha<<<1, 128, 0, stream>>>(phi, st, alpha);
  // Q = leaky(pair @ Won): (13936 x 512) @ (512 x 256), f16 out
  hmon_wmma_gemm<<<(MT_P * 16 + 7) / 8, 256, 0, stream>>>(
      pair, 512, 512, nullptr, 0, 0, WonT, 512, nullptr, Q, DN, MT_P, 16, 1);
  // e = [Q | pair] @ Weo: K = 256 + 512 = 768
  hmon_wmma_gemm<<<(MT_P * 16 + 7) / 8, 256, 0, stream>>>(
      Q, 256, 256, pair, 512, 512, WeoT, 768, ebuf, nullptr, DE, MT_P, 16, 0);

  // --- aggregation & outputs ---
  hmon_hm<<<(N_ * DE + 255) / 256, 256, 0, stream>>>(alpha, ebuf, st, hm0, hm1);
  hmon_yhat<<<1, 128, 0, stream>>>(nbuf, hm0, hm1, Wi, bi, hpmt, out);
  hmon_lort<<<1538, 128, 0, stream>>>(Wum, Wvm, Wup, Wvp, partial);
  hmon_lpol<<<1, 256, 0, stream>>>(nbuf, norms, wv, svec);
  hmon_finalize<<<1, 1, 0, stream>>>(partial, svec, out);
}
